// SelfAttentionWithBiases_57647051047593
// MI455X (gfx1250) — matrix operand order repaired and checked by reading, not compiled, not run on hardware
//
#include <hip/hip_runtime.h>
#include <hip/hip_bf16.h>
#include <math.h>

// Problem constants (match reference)
#define BATCH 8
#define LSEQ  1024
#define DMODEL 768
#define NHEAD 12
#define HDIM  64
#define TDIM  (3*DMODEL)   // 2304

typedef __attribute__((ext_vector_type(16))) _Float16 v16h;
typedef __attribute__((ext_vector_type(8)))  _Float16 v8h;
typedef __attribute__((ext_vector_type(8)))  float    v8f;

// ---------------- workspace layout (bytes) ----------------
static const size_t OFF_XN   = 0;                                   // [8192,768] f16
static const size_t SZ_XN    = (size_t)BATCH*LSEQ*DMODEL*2;
static const size_t OFF_WIN  = OFF_XN + SZ_XN;                      // [2304,768] f16
static const size_t SZ_WIN   = (size_t)TDIM*DMODEL*2;
static const size_t OFF_WOUT = OFF_WIN + SZ_WIN;                    // [768,768] f16
static const size_t SZ_WOUT  = (size_t)DMODEL*DMODEL*2;
static const size_t OFF_Q    = OFF_WOUT + SZ_WOUT;                  // [B,H,L,64] f16
static const size_t SZ_QKV   = (size_t)BATCH*NHEAD*LSEQ*HDIM*2;
static const size_t OFF_K    = OFF_Q + SZ_QKV;                      // [B,H,L,64] f16
static const size_t OFF_VT   = OFF_K + SZ_QKV;                      // [B,H,64,L] f16 (transposed)
static const size_t OFF_O    = OFF_VT + SZ_QKV;                     // [B,L,768]  f16
static const size_t SZ_O     = (size_t)BATCH*LSEQ*DMODEL*2;
static const size_t OFF_RT   = OFF_O + SZ_O;                        // [B,L,L] float2 (r,theta)
static const size_t SZ_RT    = (size_t)BATCH*LSEQ*LSEQ*8;

// ---------------- WMMA tile helpers (CDNA5 documented VGPR layouts) ----------------
// A-matrix 16x32 f16: lane m=l&15, half=l>>4; dword j: K = (j>>2)*16 + half*8 + (j&3)*2
__device__ __forceinline__ v16h load_a16x32(const _Float16* base, int lda, int kofs, int lane) {
  int m = lane & 15, half = (lane >> 4) & 1;
  const _Float16* p = base + m * lda + kofs + half * 8;
  union { v16h v; v8h h[2]; } u;
  u.h[0] = *(const v8h*)(p);        // K = half*8 + 0..7     -> dwords 0..3
  u.h[1] = *(const v8h*)(p + 16);   // K = 16 + half*8 + 0..7 -> dwords 4..7
  return u.v;
}
// B-matrix 32x16 f16: lane n=l&15, kh=l>>4; dword j: K = (j>>2)*16 + kh*8 + (j&3)*2
// Source must be K-contiguous per column n: elem (k,n) at base[n*ldb + kofs + k]
__device__ __forceinline__ v16h load_b32x16(const _Float16* base, int ldb, int kofs, int lane) {
  int n = lane & 15, kh = (lane >> 4) & 1;
  const _Float16* p = base + n * ldb + kofs + kh * 8;
  union { v16h v; v8h h[2]; } u;
  u.h[0] = *(const v8h*)(p);        // k = kh*8 + 0..7       -> dwords 0..3
  u.h[1] = *(const v8h*)(p + 16);   // k = 16 + kh*8 + 0..7  -> dwords 4..7
  return u.v;
}
__device__ __forceinline__ v8f wmma_f16(v16h a, v16h b, v8f c) {
  return __builtin_amdgcn_wmma_f32_16x16x32_f16(false, a, false, b, (short)0, c, false, false);
}
// C/D layout: lane n=l&15, mb=(l>>4)*8; dword j holds C[mb+j, n]

__device__ __forceinline__ float rmax16(float v) {
  #pragma unroll
  for (int m = 1; m < 16; m <<= 1) v = fmaxf(v, __shfl_xor(v, m, 32));
  return v;
}
__device__ __forceinline__ float rsum16(float v) {
  #pragma unroll
  for (int m = 1; m < 16; m <<= 1) v += __shfl_xor(v, m, 32);
  return v;
}

// ---------------- kernel 1: RMSNorm -> f16 ----------------
__global__ void k_rmsnorm(const float* __restrict__ x, const float* __restrict__ wn,
                          _Float16* __restrict__ xn) {
  int row = blockIdx.x;
  const float* xr = x + (size_t)row * DMODEL;
  float v[3]; float s = 0.f;
  #pragma unroll
  for (int i = 0; i < 3; i++) { v[i] = xr[threadIdx.x + 256 * i]; s += v[i] * v[i]; }
  __shared__ float red[256];
  red[threadIdx.x] = s; __syncthreads();
  for (int off = 128; off > 0; off >>= 1) {
    if (threadIdx.x < off) red[threadIdx.x] += red[threadIdx.x + off];
    __syncthreads();
  }
  float scale = rsqrtf(red[0] * (1.0f / DMODEL) + 1e-5f);
  _Float16* o = xn + (size_t)row * DMODEL;
  #pragma unroll
  for (int i = 0; i < 3; i++) {
    int c = threadIdx.x + 256 * i;
    o[c] = (_Float16)(v[i] * scale * wn[c]);
  }
}

// ---------------- kernel 2: f32 -> f16 weight convert ----------------
__global__ void k_cvt(const float* __restrict__ src, _Float16* __restrict__ dst, int n) {
  int i = blockIdx.x * 256 + threadIdx.x;
  if (i < n) dst[i] = (_Float16)src[i];
}

// ---------------- kernel 3: (r, theta) geometry table, head-independent ----------------
__global__ void k_rtheta(const float* __restrict__ pos, float2* __restrict__ rt) {
  int b = blockIdx.x >> 10, q = blockIdx.x & (LSEQ - 1);
  const float* pb = pos + (size_t)b * LSEQ * 2;
  float qy = pb[q * 2 + 0], qx = pb[q * 2 + 1];
  float2* out = rt + ((size_t)(b * LSEQ + q)) * LSEQ;
  #pragma unroll
  for (int i = 0; i < 4; i++) {
    int k = threadIdx.x + 256 * i;
    float dy = qy - pb[k * 2 + 0];
    float dx = qx - pb[k * 2 + 1];
    float th = atan2f(dx, dy);                    // arctan2(dist_x, dist_y)
    float r  = sqrtf(dy * dy + dx * dx + 1e-12f);
    out[k] = make_float2(r, th);
  }
}

// ---------------- kernel 4: QKV GEMM (WMMA), writes Q,K f16 and V transposed ----------------
// qkv[row, col] = sum_d xn[row,d] * w_in[col,d] + b_in[col]
__global__ void k_qkv(const _Float16* __restrict__ xn, const _Float16* __restrict__ win,
                      const float* __restrict__ bin,
                      _Float16* __restrict__ qf, _Float16* __restrict__ kf,
                      _Float16* __restrict__ vtf) {
  int lane = threadIdx.x & 31, w = threadIdx.x >> 5;
  int mt = blockIdx.x / 18;              // 512 M-tiles
  int nt = (blockIdx.x % 18) * 8 + w;    // 144 N-tiles, 8 per block
  const _Float16* arow = xn  + (size_t)(mt * 16) * DMODEL;
  const _Float16* brow = win + (size_t)(nt * 16) * DMODEL;
  v8f acc = {};
  for (int kt = 0; kt < DMODEL / 32; kt++) {
    v16h a  = load_a16x32(arow, DMODEL, kt * 32, lane);
    v16h bm = load_b32x16(brow, DMODEL, kt * 32, lane);
    acc = wmma_f16(a, bm, acc);
  }
  int n = lane & 15, mb = (lane >> 4) * 8;
  int col = nt * 16 + n;
  float bb = bin[col];
  int sec = col / DMODEL;
  int dcol = col - sec * DMODEL;
  int h = dcol >> 6, hd = dcol & 63;
  #pragma unroll
  for (int j = 0; j < 8; j++) {
    int row = mt * 16 + mb + j;
    int b = row >> 10, l = row & (LSEQ - 1);
    _Float16 hv = (_Float16)(acc[j] + bb);
    size_t bh = (size_t)b * NHEAD + h;
    if (sec == 0)      qf[(bh * LSEQ + l) * HDIM + hd] = hv;
    else if (sec == 1) kf[(bh * LSEQ + l) * HDIM + hd] = hv;
    else               vtf[(bh * HDIM + hd) * LSEQ + l] = hv;   // transposed for PV B-loads
  }
}

// ---------------- kernel 5: flash attention with fused analytic bias ----------------
// grid: B*H*(L/64) blocks, 128 threads (4 waves); wave w owns a 16-row q tile.
__global__ void k_attn(const _Float16* __restrict__ qf, const _Float16* __restrict__ kf,
                       const _Float16* __restrict__ vtf, const float2* __restrict__ rt,
                       const float* __restrict__ bp, _Float16* __restrict__ of) {
  __shared__ __align__(16) _Float16 pbuf[4][16][32];
  int lane = threadIdx.x & 31, w = threadIdx.x >> 5;
  int bh = blockIdx.x >> 4;            // b*H + h
  int qblk = blockIdx.x & 15;
  int b = bh / NHEAD, h = bh % NHEAD;
  int qbase = qblk * 64 + w * 16;

  // per-head bias constants
  float o_  = bp[h * 6 + 0];
  float ac  = fabsf(bp[h * 6 + 1]);
  float wph = bp[h * 6 + 2];
  float pph = bp[h * 6 + 3];
  float A   = fabsf(tanhf(bp[h * 6 + 4]));
  float am  = fabsf(bp[h * 6 + 5]);

  const _Float16* Qp = qf  + ((size_t)bh * LSEQ + qbase) * HDIM;
  const _Float16* Kp = kf  + (size_t)bh * LSEQ * HDIM;
  const _Float16* Vp = vtf + (size_t)bh * HDIM * LSEQ;
  const float2*   Rp = rt  + (size_t)(b * LSEQ) * LSEQ;

  v16h a0 = load_a16x32(Qp, HDIM, 0, lane);    // Q d=0..31
  v16h a1 = load_a16x32(Qp, HDIM, 32, lane);   // Q d=32..63

  v8f oacc[4] = {{}, {}, {}, {}};
  float mrow[8], lrow[8];
  #pragma unroll
  for (int j = 0; j < 8; j++) { mrow[j] = -1e30f; lrow[j] = 0.f; }

  int n = lane & 15, mb = (lane >> 4) * 8;

  for (int kb = 0; kb < LSEQ / 32; kb++) {
    int kbase = kb * 32;
    // S = Q K^T for 32 keys (two 16-key n-tiles)
    v8f s0 = {}, s1 = {};
    {
      const _Float16* kt0 = Kp + (size_t)(kbase + 0) * HDIM;
      const _Float16* kt1 = Kp + (size_t)(kbase + 16) * HDIM;
      s0 = wmma_f16(a0, load_b32x16(kt0, HDIM, 0, lane), s0);
      s0 = wmma_f16(a1, load_b32x16(kt0, HDIM, 32, lane), s0);
      s1 = wmma_f16(a0, load_b32x16(kt1, HDIM, 0, lane), s1);
      s1 = wmma_f16(a1, load_b32x16(kt1, HDIM, 32, lane), s1);
    }
    // scores = S/8 + bias(r,theta;head)
    float sc0[8], sc1[8];
    #pragma unroll
    for (int j = 0; j < 8; j++) {
      int qg = qbase + mb + j;
      const float2* rr = Rp + (size_t)qg * LSEQ + kbase;
      float2 g0 = rr[n], g1 = rr[16 + n];
      float t = g0.x - o_;
      float t1 = (t > 0.f) ? __expf(ac * __logf(t)) : 0.f;
      float cc = __cosf((g0.y - pph) * wph * 0.5f);
      float b2 = cc * cc;
      float t2 = 1.f - A * ((b2 > 0.f) ? __expf(am * __logf(b2)) : 0.f);
      sc0[j] = s0[j] * 0.125f + t1 * t2;
      t = g1.x - o_;
      t1 = (t > 0.f) ? __expf(ac * __logf(t)) : 0.f;
      cc = __cosf((g1.y - pph) * wph * 0.5f);
      b2 = cc * cc;
      t2 = 1.f - A * ((b2 > 0.f) ? __expf(am * __logf(b2)) : 0.f);
      sc1[j] = s1[j] * 0.125f + t1 * t2;
    }
    // online softmax update
    float scal[8];
    #pragma unroll
    for (int j = 0; j < 8; j++) {
      float mx = rmax16(fmaxf(sc0[j], sc1[j]));
      float mnew = fmaxf(mrow[j], mx);
      float sca = __expf(mrow[j] - mnew);
      float p0 = __expf(sc0[j] - mnew);
      float p1 = __expf(sc1[j] - mnew);
      lrow[j] = lrow[j] * sca + rsum16(p0 + p1);
      mrow[j] = mnew;
      scal[j] = sca;
      pbuf[w][mb + j][n] = (_Float16)p0;        // C-layout -> LDS
      pbuf[w][mb + j][16 + n] = (_Float16)p1;
    }
    #pragma unroll
    for (int j = 0; j < 8; j++) {
      float sca = scal[j];
      oacc[0][j] *= sca; oacc[1][j] *= sca; oacc[2][j] *= sca; oacc[3][j] *= sca;
    }
    // P (16x32, A-layout from LDS) times V (32 keys x 64 dims, via transposed V)
    v16h pa = load_a16x32(&pbuf[w][0][0], 32, 0, lane);
    #pragma unroll
    for (int t = 0; t < 4; t++) {
      v16h bv = load_b32x16(Vp + (size_t)(t * 16) * LSEQ, LSEQ, kbase, lane);
      oacc[t] = wmma_f16(pa, bv, oacc[t]);
    }
  }
  // normalize and store O into [B, L, H*64] layout (ready for out-proj A-matrix)
  float linv[8];
  #pragma unroll
  for (int j = 0; j < 8; j++) linv[j] = 1.0f / lrow[j];
  #pragma unroll
  for (int t = 0; t < 4; t++) {
    #pragma unroll
    for (int j = 0; j < 8; j++) {
      int qg = qbase + mb + j;
      of[(size_t)(b * LSEQ + qg) * DMODEL + h * HDIM + t * 16 + n] =
          (_Float16)(oacc[t][j] * linv[j]);
    }
  }
}

// ---------------- kernel 6: output projection GEMM (WMMA), f32 out ----------------
__global__ void k_out(const _Float16* __restrict__ of, const _Float16* __restrict__ wout,
                      const float* __restrict__ bout, float* __restrict__ out) {
  int lane = threadIdx.x & 31, w = threadIdx.x >> 5;
  int mt = blockIdx.x / 6;               // 512 M-tiles
  int nt = (blockIdx.x % 6) * 8 + w;     // 48 N-tiles
  const _Float16* arow = of   + (size_t)(mt * 16) * DMODEL;
  const _Float16* brow = wout + (size_t)(nt * 16) * DMODEL;
  v8f acc = {};
  for (int kt = 0; kt < DMODEL / 32; kt++) {
    v16h a  = load_a16x32(arow, DMODEL, kt * 32, lane);
    v16h bm = load_b32x16(brow, DMODEL, kt * 32, lane);
    acc = wmma_f16(a, bm, acc);
  }
  int n = lane & 15, mb = (lane >> 4) * 8;
  int col = nt * 16 + n;
  float bb = bout[col];
  #pragma unroll
  for (int j = 0; j < 8; j++) {
    int row = mt * 16 + mb + j;
    out[(size_t)row * DMODEL + col] = acc[j] + bb;
  }
}

// ---------------- host launcher ----------------
extern "C" void kernel_launch(void* const* d_in, const int* in_sizes, int n_in,
                              void* d_out, int out_size, void* d_ws, size_t ws_size,
                              hipStream_t stream) {
  (void)in_sizes; (void)n_in; (void)out_size; (void)ws_size;
  const float* x      = (const float*)d_in[0];
  const float* pos    = (const float*)d_in[1];
  const float* w_norm = (const float*)d_in[2];
  const float* w_in   = (const float*)d_in[3];
  const float* b_in   = (const float*)d_in[4];
  const float* w_out  = (const float*)d_in[5];
  const float* b_out  = (const float*)d_in[6];
  const float* bparam = (const float*)d_in[7];
  float* out = (float*)d_out;

  char* ws = (char*)d_ws;
  _Float16* xn   = (_Float16*)(ws + OFF_XN);
  _Float16* winh = (_Float16*)(ws + OFF_WIN);
  _Float16* wouh = (_Float16*)(ws + OFF_WOUT);
  _Float16* qf   = (_Float16*)(ws + OFF_Q);
  _Float16* kf   = (_Float16*)(ws + OFF_K);
  _Float16* vtf  = (_Float16*)(ws + OFF_VT);
  _Float16* of   = (_Float16*)(ws + OFF_O);
  float2*   rt   = (float2*)(ws + OFF_RT);

  k_rmsnorm<<<BATCH * LSEQ, 256, 0, stream>>>(x, w_norm, xn);
  k_cvt<<<(TDIM * DMODEL + 255) / 256, 256, 0, stream>>>(w_in, winh, TDIM * DMODEL);
  k_cvt<<<(DMODEL * DMODEL + 255) / 256, 256, 0, stream>>>(w_out, wouh, DMODEL * DMODEL);
  k_rtheta<<<BATCH * LSEQ, 256, 0, stream>>>(pos, rt);
  k_qkv<<<512 * 18, 256, 0, stream>>>(xn, winh, b_in, qf, kf, vtf);
  k_attn<<<BATCH * NHEAD * (LSEQ / 64), 128, 0, stream>>>(qf, kf, vtf, rt, bparam, of);
  k_out<<<512 * 6, 256, 0, stream>>>(of, wouh, b_out, out);
}
